// MaskGuidedContrastiveLoss_72456098284079
// MI455X (gfx1250) — compile-verified, matrix operand order checked by reading, not run on hardware
//
#include <hip/hip_runtime.h>
#include <hip/hip_bf16.h>
#include <math.h>
#include <stdint.h>

typedef __attribute__((ext_vector_type(16))) _Float16 v16h;
typedef __attribute__((ext_vector_type(8)))  _Float16 h8;
typedef __attribute__((ext_vector_type(8)))  float    v8f;

#define N_TOT    16384
#define HW       4096
#define D        64
#define INV_TEMP 14.285714285714286f   /* 1/0.07 */
#define M0F      INV_TEMP              /* static LSE max bound: |sim|/TEMP <= 1/TEMP */
#define NEG_INF  (-__builtin_inff())

/* workspace byte offsets */
#define OFF_COUNTS 0
#define OFF_POSIDX 64
#define OFF_NEGIDX (OFF_POSIDX + N_TOT * 4)
#define OFF_POSM   (OFF_NEGIDX + N_TOT * 4)
#define OFF_NEGM   (OFF_POSM + N_TOT * D * 2)
#define OFF_TERMS  (OFF_NEGM + N_TOT * D * 2)

/* ---------------- kernel 1: deterministic index compaction ---------------- */
__global__ __launch_bounds__(1024) void build_indices(const float* __restrict__ mask,
                                                      int* __restrict__ pos_idx,
                                                      int* __restrict__ neg_idx,
                                                      int* __restrict__ counts) {
    __shared__ int scan[1024];
    __shared__ int basePos, baseNeg;
    const int tid = threadIdx.x;
    if (tid == 0) { basePos = 0; baseNeg = 0; }
    __syncthreads();
    for (int chunk = 0; chunk < N_TOT / 1024; ++chunk) {
        const int i = chunk * 1024 + tid;
        const int bit = (mask[i] > 0.5f) ? 1 : 0;
        scan[tid] = bit;
        __syncthreads();
        for (int off = 1; off < 1024; off <<= 1) {
            int v = (tid >= off) ? scan[tid - off] : 0;
            __syncthreads();
            scan[tid] += v;
            __syncthreads();
        }
        const int incl = scan[tid];
        const int chunkPos = scan[1023];
        if (bit) pos_idx[basePos + incl - 1] = i;
        else     neg_idx[baseNeg + (tid + 1 - incl) - 1] = i;
        __syncthreads();
        if (tid == 0) { basePos += chunkPos; baseNeg += 1024 - chunkPos; }
        __syncthreads();
    }
    if (tid == 0) {
        const int nPos = basePos, nNegTot = baseNeg;
        int nNegUse = nNegTot >> 1;                 /* NEG_RATIO = 0.5 */
        if (nNegUse < 1) nNegUse = 1;
        if (nNegUse > nNegTot) nNegUse = nNegTot;   /* nNegTot==0 -> 0 */
        const int nAnchor = nPos >> 1;              /* ANCHOR_RATIO = 0.5 */
        counts[0] = nPos; counts[1] = nNegTot; counts[2] = nNegUse; counts[3] = nAnchor;
    }
}

/* -------- kernel 2: gather (b,d,h,w)->(row,d), L2-normalize, cast f16 ----- */
__global__ __launch_bounds__(256) void gather_normalize(const float* __restrict__ z,
                                                        const int* __restrict__ idx_list,
                                                        const int* __restrict__ counts,
                                                        int which,
                                                        _Float16* __restrict__ outM) {
    const int wave = threadIdx.x >> 5, lane = threadIdx.x & 31;
    const int r = blockIdx.x * (blockDim.x >> 5) + wave;
    const int count = (which == 0) ? counts[0] : counts[2];
    if (r >= count) return;
    const int idx = idx_list[r];
    const int b = idx >> 12;        /* idx / HW */
    const int p = idx & (HW - 1);   /* idx % HW */
    const float* zb = z + (size_t)b * D * HW + p;
    const float v0 = zb[(size_t)lane * HW];
    const float v1 = zb[(size_t)(lane + 32) * HW];
    float ss = v0 * v0 + v1 * v1;
#pragma unroll
    for (int off = 1; off < 32; off <<= 1) ss += __shfl_xor(ss, off, 32);
    const float inv = 1.0f / fmaxf(sqrtf(ss), 1e-12f);
    outM[(size_t)r * D + lane]      = (_Float16)(v0 * inv);
    outM[(size_t)r * D + lane + 32] = (_Float16)(v1 * inv);
}

/* A/B 16x32 f16 fragment pair (K 0..31, 32..63) per ISA layout:
   lane L: row = L&15, kb = (L>=16)*8; K chunks {kb..kb+7} and {16+kb..16+kb+7}. */
__device__ __forceinline__ void frag_from_row(const _Float16* rowBase,
                                              int lane, v16h& f0, v16h& f1) {
    const h8* rp = (const h8*)(rowBase + (size_t)(lane & 15) * D);
    const int cb = lane >> 4;          /* 0/1: 16B chunk inside each 32B half */
    h8 q0 = rp[0 + cb], q1 = rp[2 + cb], q2 = rp[4 + cb], q3 = rp[6 + cb];
    f0 = __builtin_shufflevector(q0, q1, 0,1,2,3,4,5,6,7,8,9,10,11,12,13,14,15);
    f1 = __builtin_shufflevector(q2, q3, 0,1,2,3,4,5,6,7,8,9,10,11,12,13,14,15);
}

/* Async-stage a 32x64 f16 chunk (4KB) into LDS: 128 threads x 2 x 16B.
   Tracked on ASYNCcnt; consumer does s_wait_asynccnt + barrier.            */
__device__ __forceinline__ void stage_chunk_async(const _Float16* __restrict__ mat,
                                                  int count, int ck,
                                                  _Float16* ldsDst, int tid) {
#pragma unroll
    for (int q = 0; q < 2; ++q) {
        const int c = tid * 2 + q;          /* 0..255 16B chunks */
        const int row = c >> 3, ch = c & 7;
        int r = ck * 32 + row; if (r >= count) r = count - 1;
        const _Float16* g = mat + (size_t)r * D + ch * 8;
        const unsigned lds = (unsigned)(uintptr_t)(ldsDst + row * D + ch * 8);
        const unsigned long long ga = (unsigned long long)(uintptr_t)g;
        asm volatile("global_load_async_to_lds_b128 %0, %1, off"
                     :: "v"(lds), "v"(ga) : "memory");
    }
}

__device__ __forceinline__ void accum_max(const v8f& c, float* pmax, int tIdx, int diagT,
                                          int lastT, int count, int aBase, int m, int cb) {
    if ((tIdx != diagT) && (tIdx != lastT)) {
#pragma unroll
        for (int v = 0; v < 8; ++v) pmax[v] = fmaxf(pmax[v], c[v]);
    } else {
        const int gcol = tIdx * 16 + m;
#pragma unroll
        for (int v = 0; v < 8; ++v) {
            const int mrow = v + (cb << 3);
            float val = c[v];
            if (gcol >= count || gcol == aBase + mrow) val = NEG_INF;
            pmax[v] = fmaxf(pmax[v], val);
        }
    }
}

__device__ __forceinline__ void accum_lse(const v8f& c, float* nsum, int tIdx,
                                          int lastT, int count, int m) {
    if (tIdx != lastT) {
#pragma unroll
        for (int v = 0; v < 8; ++v)
            nsum[v] += __expf(__builtin_fmaf(c[v], INV_TEMP, -M0F));
    } else {
        const int gcol = tIdx * 16 + m;
#pragma unroll
        for (int v = 0; v < 8; ++v)
            if (gcol < count)
                nsum[v] += __expf(__builtin_fmaf(c[v], INV_TEMP, -M0F));
    }
}

__device__ __forceinline__ void reduce_emit(const float* pmax, const float* nsum,
                                            int aBase, int nAnchor,
                                            float* __restrict__ terms, int lane) {
    const bool owner = ((lane & 8) == 0);   /* lanes 0-7 rows 0-7; 16-23 rows 8-15 */
    const int ownV = lane & 7;
    const int ownRow = ownV + ((lane >> 4) << 3);
    float rowPM = NEG_INF, rowNS = 0.0f;
#pragma unroll
    for (int v = 0; v < 8; ++v) {
        float pm = pmax[v], ns = nsum[v];
#pragma unroll
        for (int off = 1; off < 16; off <<= 1) {
            pm = fmaxf(pm, __shfl_xor(pm, off, 32));
            ns += __shfl_xor(ns, off, 32);
        }
        if (owner && v == ownV) { rowPM = pm; rowNS = ns; }
    }
    if (owner) {
        const int a = aBase + ownRow;
        const float ps = rowPM * INV_TEMP;  /* scale after max (monotone) */
        const float lneg = (rowNS > 0.0f) ? (M0F + __logf(rowNS)) : NEG_INF;
        const float mm = fmaxf(ps, lneg);
        float term;
        if (mm == NEG_INF) term = 0.0f;
        else {
            const float ep = (ps   > NEG_INF) ? __expf(ps - mm)   : 0.0f;
            const float en = (lneg > NEG_INF) ? __expf(lneg - mm) : 0.0f;
            term = -ps + mm + __logf(ep + en);
        }
        if (a < nAnchor) terms[a] = term;
    }
}

/* ---- kernel 3: fused WMMA similarity + row max + fixed-max logsumexp -----
   128 threads = 4 waves; each wave owns 32 anchors (2 A tiles); block = 128
   anchors. B columns staged 32 at a time into double-buffered LDS via
   async-to-LDS; 8 WMMAs per wave per pipeline step.                         */
__global__ __launch_bounds__(128) void contrastive_main(const _Float16* __restrict__ posM,
                                                        const _Float16* __restrict__ negM,
                                                        const int* __restrict__ counts,
                                                        float* __restrict__ terms) {
    __shared__ _Float16 tile[2][32 * D];   /* double-buffered 32x64 f16, 8KB */
    const int nPos = counts[0];
    const int nNeg = counts[2];
    const int nAnchor = counts[3];
    const int blkA = blockIdx.x * 128;
    if (blkA >= nAnchor) return;           /* uniform per block */

    const int tid = threadIdx.x;
    const int wave = tid >> 5, lane = tid & 31;
    const int a0 = blkA + wave * 32;       /* this wave's 32 anchors */
    const int m = lane & 15, cb = lane >> 4;

    /* anchors = first nAnchor positive rows; 2 A-fragment pairs */
    int ar0 = a0 + m;      if (ar0 >= nPos) ar0 = nPos - 1;
    int ar1 = a0 + 16 + m; if (ar1 >= nPos) ar1 = nPos - 1;
    v16h af0, af1, ag0, ag1;
    frag_from_row(posM + (size_t)(ar0 - m) * D, lane, af0, af1);
    frag_from_row(posM + (size_t)(ar1 - m) * D, lane, ag0, ag1);

    float pmaxA[8], pmaxB[8], nsumA[8], nsumB[8];
#pragma unroll
    for (int v = 0; v < 8; ++v) {
        pmaxA[v] = NEG_INF; pmaxB[v] = NEG_INF;
        nsumA[v] = 0.0f;    nsumB[v] = 0.0f;
    }

    /* ---------- positives: per-lane raw max, diag/tail tiles special ------ */
    {
        const int nT16 = (nPos + 15) >> 4;
        const int nC = (nPos + 31) >> 5;
        const int diagT0 = a0 >> 4, diagT1 = diagT0 + 1;
        stage_chunk_async(posM, nPos, 0, &tile[0][0], tid);
        for (int ck = 0; ck < nC; ++ck) {
            asm volatile("s_wait_asynccnt 0x0" ::: "memory");
            __syncthreads();
            if (ck + 1 < nC) stage_chunk_async(posM, nPos, ck + 1, &tile[(ck + 1) & 1][0], tid);
            const _Float16* tb = &tile[ck & 1][0];
#pragma unroll
            for (int s = 0; s < 2; ++s) {
                const int tIdx = ck * 2 + s;
                if (tIdx < nT16) {
                    v16h b0, b1;
                    frag_from_row(tb + s * 16 * D, lane, b0, b1);
                    v8f c = {};
                    c = __builtin_amdgcn_wmma_f32_16x16x32_f16(false, af0, false, b0, (short)0, c, false, false);
                    c = __builtin_amdgcn_wmma_f32_16x16x32_f16(false, af1, false, b1, (short)0, c, false, false);
                    accum_max(c, pmaxA, tIdx, diagT0, nT16 - 1, nPos, a0, m, cb);
                    v8f d = {};
                    d = __builtin_amdgcn_wmma_f32_16x16x32_f16(false, ag0, false, b0, (short)0, d, false, false);
                    d = __builtin_amdgcn_wmma_f32_16x16x32_f16(false, ag1, false, b1, (short)0, d, false, false);
                    accum_max(d, pmaxB, tIdx, diagT1, nT16 - 1, nPos, a0 + 16, m, cb);
                }
            }
        }
    }
    __syncthreads();   /* protect tile[] before neg loop restages buffer 0 */

    /* ---------- negatives: fixed-max logsumexp accumulation --------------- */
    {
        const int nT16 = (nNeg + 15) >> 4;
        const int nC = (nNeg + 31) >> 5;
        if (nC > 0) {
            stage_chunk_async(negM, nNeg, 0, &tile[0][0], tid);
            for (int ck = 0; ck < nC; ++ck) {
                asm volatile("s_wait_asynccnt 0x0" ::: "memory");
                __syncthreads();
                if (ck + 1 < nC) stage_chunk_async(negM, nNeg, ck + 1, &tile[(ck + 1) & 1][0], tid);
                const _Float16* tb = &tile[ck & 1][0];
#pragma unroll
                for (int s = 0; s < 2; ++s) {
                    const int tIdx = ck * 2 + s;
                    if (tIdx < nT16) {
                        v16h b0, b1;
                        frag_from_row(tb + s * 16 * D, lane, b0, b1);
                        v8f c = {};
                        c = __builtin_amdgcn_wmma_f32_16x16x32_f16(false, af0, false, b0, (short)0, c, false, false);
                        c = __builtin_amdgcn_wmma_f32_16x16x32_f16(false, af1, false, b1, (short)0, c, false, false);
                        accum_lse(c, nsumA, tIdx, nT16 - 1, nNeg, m);
                        v8f d = {};
                        d = __builtin_amdgcn_wmma_f32_16x16x32_f16(false, ag0, false, b0, (short)0, d, false, false);
                        d = __builtin_amdgcn_wmma_f32_16x16x32_f16(false, ag1, false, b1, (short)0, d, false, false);
                        accum_lse(d, nsumB, tIdx, nT16 - 1, nNeg, m);
                    }
                }
            }
        }
    }

    /* ---------- one-shot 16-lane group reduction + term emit -------------- */
    reduce_emit(pmaxA, nsumA, a0,      nAnchor, terms, lane);
    reduce_emit(pmaxB, nsumB, a0 + 16, nAnchor, terms, lane);
}

/* ---------------- kernel 4: mean reduce + degenerate case ---------------- */
__global__ __launch_bounds__(256) void finalize(const float* __restrict__ terms,
                                                const int* __restrict__ counts,
                                                float* __restrict__ out) {
    __shared__ float red[256];
    const int nPos = counts[0], nAnchor = counts[3];
    float s = 0.0f;
    for (int i = threadIdx.x; i < nAnchor; i += 256) s += terms[i];
    red[threadIdx.x] = s;
    __syncthreads();
    for (int off = 128; off > 0; off >>= 1) {
        if (threadIdx.x < off) red[threadIdx.x] += red[threadIdx.x + off];
        __syncthreads();
    }
    if (threadIdx.x == 0)
        out[0] = (nPos < 10) ? 0.01f : red[0] / (float)nAnchor;
}

extern "C" void kernel_launch(void* const* d_in, const int* in_sizes, int n_in,
                              void* d_out, int out_size, void* d_ws, size_t ws_size,
                              hipStream_t stream) {
    const float* z    = (const float*)d_in[0];
    const float* mask = (const float*)d_in[1];
    float* out = (float*)d_out;
    char* ws = (char*)d_ws;

    int*      counts  = (int*)(ws + OFF_COUNTS);
    int*      pos_idx = (int*)(ws + OFF_POSIDX);
    int*      neg_idx = (int*)(ws + OFF_NEGIDX);
    _Float16* posM    = (_Float16*)(ws + OFF_POSM);
    _Float16* negM    = (_Float16*)(ws + OFF_NEGM);
    float*    terms   = (float*)(ws + OFF_TERMS);

    hipLaunchKernelGGL(build_indices, dim3(1), dim3(1024), 0, stream,
                       mask, pos_idx, neg_idx, counts);
    hipLaunchKernelGGL(gather_normalize, dim3(N_TOT / 8), dim3(256), 0, stream,
                       z, pos_idx, counts, 0, posM);
    hipLaunchKernelGGL(gather_normalize, dim3(N_TOT / 8), dim3(256), 0, stream,
                       z, neg_idx, counts, 1, negM);
    /* max anchors = 8192 -> 64 blocks x 128 anchors (4 waves x 32) */
    hipLaunchKernelGGL(contrastive_main, dim3(64), dim3(128), 0, stream,
                       posM, negM, counts, terms);
    hipLaunchKernelGGL(finalize, dim3(1), dim3(256), 0, stream,
                       terms, counts, out);
}
